// RephineLayer_45071386804961
// MI455X (gfx1250) — matrix operand order repaired and checked by reading, not compiled, not run on hardware
//
#include <hip/hip_runtime.h>

// ---------------- problem constants ----------------
#define BG      1024
#define NNODES  128
#define EEDGES  256
#define VTOT    (BG * NNODES)     // 131072
#define ETOT    (BG * EEDGES)     // 262144
#define NFEAT   128
#define FHID    64
#define NFILT   8
#define OUTD    64
#define NPROB   (NFILT * BG)      // 8192

typedef _Float16 f16;
typedef f16   v16h __attribute__((ext_vector_type(16)));
typedef f16   v8h  __attribute__((ext_vector_type(8)));
typedef float v8f  __attribute__((ext_vector_type(8)));
typedef float v4f  __attribute__((ext_vector_type(4)));

// =====================================================================
// Kernel 1: filtration MLP.  rows x NFEAT -> relu GEMM (WMMA) -> 64
//           -> small GEMM -> sigmoid -> 8 outputs per row.
// A and B tiles are staged in LDS pre-swizzled into a contiguous-k (k = 16*hl + elem) WMMA
// fragment layout, so each lane fragment is a contiguous 32B v16h
// (2x ds_load_b128) instead of 16 scalar ds_load_u16.
// GATHER=false : rows are vertices.  GATHER=true : rows are edges.
// Block = 256 threads (8 waves), 128 rows per block.
// =====================================================================
template <bool GATHER>
__global__ void __launch_bounds__(256)
filt_kernel(const float* __restrict__ x,
            const int*   __restrict__ e0,
            const int*   __restrict__ e1,
            const float* __restrict__ W1,   // [128,64]
            const float* __restrict__ b1,   // [64]
            const float* __restrict__ W2,   // [64,8]
            const float* __restrict__ b2,   // [8]
            float*       __restrict__ fout) // [rows,8]
{
    // As: [mtile=8][kk=4][lane=32] v16h  = 32 KiB, reused as Hs f32[128][64]
    // Bs: [ntile=4][kk=4][lane=32] v16h  = 16 KiB
    __shared__ v16h As[8][4][32];
    __shared__ v16h Bs[4][4][32];
    __shared__ float b1s[64];
    __shared__ float W2s[64][8];
    __shared__ float b2s[8];
    float (*Hs)[64] = (float(*)[64])As;

    const int t    = threadIdx.x;
    const int row0 = blockIdx.x * 128;

    // ---- stage B = W1 into WMMA fragment layout ----
    // W1[k][n]: lane = (n&15) + 16*(kc>>4), elem = kc&15   (kc = k&31)
    {
        f16* bp = (f16*)Bs;
        for (int i = t; i < 128 * 64; i += 256) {
            int k = i >> 6, n = i & 63;
            int kk = k >> 5, kc = k & 31;
            bp[(((n >> 4) * 4 + kk) * 32 + (n & 15) + 16 * (kc >> 4)) * 16 + (kc & 15)]
                = (f16)W1[i];
        }
    }
    if (t < 64) b1s[t] = b1[t];
    for (int i = t; i < 64 * 8; i += 256) W2s[i >> 3][i & 7] = W2[i];
    if (t < 8)  b2s[t] = b2[t];

    // ---- stage A: 8 consecutive columns per thread -> one 16B LDS store ----
    // A elem: lane = (r&15) + 16*(kc>>4), elem = kc&15
    {
        f16* ap = (f16*)As;
        for (int gidx = t; gidx < 2048; gidx += 256) {
            int r  = gidx >> 4;
            int cg = (gidx & 15) * 8;
            v4f lo, hi;
            if (GATHER) {
                int et = row0 + r;
                const float* p0 = x + (long)e0[et] * NFEAT + cg;
                const float* p1 = x + (long)e1[et] * NFEAT + cg;
                lo = *(const v4f*)p0 + *(const v4f*)p1;
                hi = *(const v4f*)(p0 + 4) + *(const v4f*)(p1 + 4);
            } else {
                const float* p0 = x + (long)(row0 + r) * NFEAT + cg;
                lo = *(const v4f*)p0;
                hi = *(const v4f*)(p0 + 4);
            }
            v8h hbuf;
#pragma unroll
            for (int q = 0; q < 4; q++) { hbuf[q] = (f16)lo[q]; hbuf[q + 4] = (f16)hi[q]; }
            int kk  = cg >> 5, kc0 = cg & 31;
            int hl  = kc0 >> 4;
            int eb  = kc0 & 8;
            *(v8h*)&ap[(((r >> 4) * 4 + kk) * 32 + (r & 15) + 16 * hl) * 16 + eb] = hbuf;
        }
    }
    __syncthreads();

    const int wave = t >> 5;
    const int lane = t & 31;
    const int m0   = wave * 16;       // each wave owns 16 rows, all 4 N-tiles
    const int mrow = lane & 15;
    const int hl   = lane >> 4;

    // A fragments for 4 K-steps: contiguous v16h per lane
    v16h a[4];
#pragma unroll
    for (int kk = 0; kk < 4; kk++) a[kk] = As[wave][kk][lane];

    v8f cacc[4];
#pragma unroll
    for (int n = 0; n < 4; n++) {
        v8f c = {};
#pragma unroll
        for (int kk = 0; kk < 4; kk++) {
            v16h bf = Bs[n][kk][lane];
            c = __builtin_amdgcn_wmma_f32_16x16x32_f16(
                    false, a[kk], false, bf, (short)0, c, false, false);
        }
        cacc[n] = c;
    }
    __syncthreads();                  // done reading As/Bs

    // write relu(C + b1) into Hs (C layout: vgpr j -> M=j+8*hl, lane -> N)
#pragma unroll
    for (int n = 0; n < 4; n++) {
#pragma unroll
        for (int j = 0; j < 8; j++) {
            int mm = m0 + j + 8 * hl;
            int cc = n * 16 + mrow;
            Hs[mm][cc] = fmaxf(cacc[n][j] + b1s[cc], 0.f);
        }
    }
    __syncthreads();

    // GEMM2: [128,64]@[64,8] + sigmoid. thread -> (row, 4 outputs)
    {
        int r = t >> 1, ob = (t & 1) * 4;
        float s0 = b2s[ob], s1 = b2s[ob + 1], s2 = b2s[ob + 2], s3 = b2s[ob + 3];
        for (int j = 0; j < 64; j++) {
            float hv = Hs[r][j];
            s0 += hv * W2s[j][ob];     s1 += hv * W2s[j][ob + 1];
            s2 += hv * W2s[j][ob + 2]; s3 += hv * W2s[j][ob + 3];
        }
        float* o = &fout[(long)(row0 + r) * NFILT + ob];
        o[0] = 1.f / (1.f + expf(-s0));
        o[1] = 1.f / (1.f + expf(-s1));
        o[2] = 1.f / (1.f + expf(-s2));
        o[3] = 1.f / (1.f + expf(-s3));
    }
}

// =====================================================================
// gamma: per-vertex min incident edge filtration (1.0 if isolated)
// =====================================================================
__global__ void gamma_init(float* gam) {
    gam[blockIdx.x * 256 + threadIdx.x] = 1.0f;
}
__global__ void gamma_scatter(const float* __restrict__ fe,
                              const int* __restrict__ e0,
                              const int* __restrict__ e1,
                              unsigned int* __restrict__ gamU) {
    int i  = blockIdx.x * 256 + threadIdx.x;   // i = et*8 + f
    int et = i >> 3, f = i & 7;
    unsigned int bits = __float_as_uint(fe[i]); // fe>0 -> bit order == float order
    atomicMin(&gamU[(long)e0[et] * NFILT + f], bits);
    atomicMin(&gamU[(long)e1[et] * NFILT + f], bits);
}

// =====================================================================
// Sort: stable ascending argsort of 256 edge values per (f,b) problem.
// Bitonic on 64-bit keys (bits(fe)<<32 | e).  order stored transposed
// [k][p] for coalesced UF reads.
// =====================================================================
__global__ void __launch_bounds__(256)
sort_kernel(const float* __restrict__ fe, unsigned short* __restrict__ order_t) {
    int p = blockIdx.x;              // f*1024 + b
    int f = p >> 10, b = p & 1023;
    int t = threadIdx.x;
    __shared__ unsigned long long keys[256];
    unsigned int bits = __float_as_uint(fe[(long)(b * EEDGES + t) * NFILT + f]);
    keys[t] = ((unsigned long long)bits << 32) | (unsigned int)t;
    __syncthreads();
    for (int ksz = 2; ksz <= 256; ksz <<= 1) {
        for (int j = ksz >> 1; j > 0; j >>= 1) {
            int ixj = t ^ j;
            if (ixj > t) {
                bool up = ((t & ksz) == 0);
                unsigned long long ka = keys[t], kb = keys[ixj];
                if ((ka > kb) == up) { keys[t] = kb; keys[ixj] = ka; }
            }
            __syncthreads();
        }
    }
    order_t[t * NPROB + p] = (unsigned short)(keys[t] & 0xFFFFu);
}

// =====================================================================
// Union-find elder-rule persistence: one problem per lane (wave32-
// friendly: 32 serial UF loops run in lockstep on different data).
// =====================================================================
__global__ void __launch_bounds__(64)
uf_kernel(const float* __restrict__ fv,
          const int*   __restrict__ e0g,
          const int*   __restrict__ e1g,
          const unsigned short* __restrict__ order_t,
          short*         __restrict__ death,   // [NPROB][128]
          unsigned char* __restrict__ cyc)     // [NPROB][256]
{
    int t = threadIdx.x;
    int p = blockIdx.x * 64 + t;
    int f = p >> 10, b = p & 1023;
    __shared__ unsigned char parent[64][128];
    __shared__ float         fvs[64][128];

    for (int n = 0; n < 128; n++) {
        parent[t][n] = (unsigned char)n;
        fvs[t][n]    = fv[(long)(b * NNODES + n) * NFILT + f];
        death[(long)p * 128 + n] = -1;
    }
    const int ebase = b * EEDGES;
    const int vbase = b * NNODES;
    for (int k = 0; k < 256; k++) {
        int e = order_t[k * NPROB + p];
        int u = e0g[ebase + e] - vbase;
        int w = e1g[ebase + e] - vbase;
        int ru = u;
        while (parent[t][ru] != ru) { parent[t][ru] = parent[t][parent[t][ru]]; ru = parent[t][ru]; }
        int rw = w;
        while (parent[t][rw] != rw) { parent[t][rw] = parent[t][parent[t][rw]]; rw = parent[t][rw]; }
        if (ru != rw) {
            float fu = fvs[t][ru], fw = fvs[t][rw];
            bool uy    = (fu > fw) || ((fu == fw) && (ru > rw)); // younger
            int young  = uy ? ru : rw;
            int eld    = uy ? rw : ru;
            parent[t][young] = (unsigned char)eld;
            death[(long)p * 128 + young] = (short)e;
            cyc[(long)p * 256 + e] = 0;
        } else {
            cyc[(long)p * 256 + e] = 1;
        }
    }
}

// =====================================================================
// Per-graph pooling: pooled0 (RePHINE dim-0 tuples @ Wds0, relu, sum)
// + pooled1 (cycle edges: relu(A0 + birth*A1), sum over e, mean over f)
// =====================================================================
__global__ void __launch_bounds__(256)
pool_kernel(const float* __restrict__ fv, const float* __restrict__ fe,
            const float* __restrict__ gam,
            const short* __restrict__ death, const unsigned char* __restrict__ cyc,
            const float* __restrict__ Wds0, const float* __restrict__ bds0,
            const float* __restrict__ Wds1, const float* __restrict__ bds1,
            float* __restrict__ h)
{
    int b = blockIdx.x, t = threadIdx.x;
    __shared__ float x0s[128][32];
    __shared__ float W0s[32][64];
    __shared__ float birth_s[2048];
    __shared__ unsigned char cyc_s[2048];
    __shared__ float A0[64], A1[64];
    __shared__ float red[2][4][64];

    for (int i = t; i < 32 * 64; i += 256) W0s[i >> 6][i & 63] = Wds0[i];
    if (t < 64) { A0[t] = Wds1[64 + t] + bds1[t]; A1[t] = Wds1[128 + t] + Wds1[192 + t]; }

    for (int i = t; i < 1024; i += 256) {          // (n,f) tuples
        int n = i >> 3, f = i & 7;
        int p = f * BG + b;
        int de = death[(long)p * 128 + n];
        float d = (de >= 0) ? fe[(long)(b * EEDGES + de) * NFILT + f] : 1.0f;
        long v = (long)(b * NNODES + n) * NFILT + f;
        x0s[n][4 * f + 0] = 0.f;
        x0s[n][4 * f + 1] = d;
        x0s[n][4 * f + 2] = gam[v];
        x0s[n][4 * f + 3] = fv[v];
    }
    for (int i = t; i < 2048; i += 256) {          // (f,e) tuples
        int f = i >> 8, e = i & 255;
        cyc_s[i]   = cyc[(long)(f * BG + b) * 256 + e];
        birth_s[i] = fe[(long)(b * EEDGES + e) * NFILT + f];
    }
    __syncthreads();

    int o = t & 63, g = t >> 6;
    float bo = bds0[o];
    float acc0 = 0.f;
    for (int n = 32 * g; n < 32 * g + 32; n++) {
        float s = bo;
#pragma unroll
        for (int j = 0; j < 32; j++) s += x0s[n][j] * W0s[j][o];
        acc0 += fmaxf(s, 0.f);
    }
    float acc1 = 0.f;
    float a0 = A0[o], a1 = A1[o];
    for (int q = 512 * g; q < 512 * g + 512; q++) {
        if (cyc_s[q]) acc1 += fmaxf(a0 + birth_s[q] * a1, 0.f);
    }
    red[0][g][o] = acc0;
    red[1][g][o] = acc1;
    __syncthreads();
    if (t < 64) {
        float s0 = red[0][0][t] + red[0][1][t] + red[0][2][t] + red[0][3][t];
        float s1 = red[1][0][t] + red[1][1][t] + red[1][2][t] + red[1][3][t];
        h[(long)b * OUTD + t] = s0 + s1 * (1.0f / NFILT);
    }
}

// =====================================================================
// Batch statistics (training BatchNorm, ddof=0) over 1024 rows
// =====================================================================
__global__ void __launch_bounds__(256)
stats_kernel(const float* __restrict__ h, float* __restrict__ mu, float* __restrict__ var) {
    int t = threadIdx.x, o = t & 63, g = t >> 6;
    float s = 0.f, s2 = 0.f;
    for (int r = g; r < BG; r += 4) {
        float v = h[(long)r * OUTD + o];
        s += v; s2 += v * v;
    }
    __shared__ float rs[4][64], rs2[4][64];
    rs[g][o] = s; rs2[g][o] = s2;
    __syncthreads();
    if (t < 64) {
        float S  = rs[0][t] + rs[1][t] + rs[2][t] + rs[3][t];
        float S2 = rs2[0][t] + rs2[1][t] + rs2[2][t] + rs2[3][t];
        float m = S * (1.0f / BG);
        mu[t]  = m;
        var[t] = S2 * (1.0f / BG) - m * m;
    }
}

// =====================================================================
// Final: BN affine -> relu(hn@Wo1+bo1)@Wo2+bo2
// =====================================================================
__global__ void __launch_bounds__(64)
out_kernel(const float* __restrict__ h, const float* __restrict__ mu,
           const float* __restrict__ var, const float* __restrict__ gw,
           const float* __restrict__ bw,  const float* __restrict__ Wo1,
           const float* __restrict__ bo1, const float* __restrict__ Wo2,
           const float* __restrict__ bo2, float* __restrict__ out) {
    int r = blockIdx.x, t = threadIdx.x;
    __shared__ float hn[64], t1[64];
    float v = h[(long)r * OUTD + t];
    hn[t] = (v - mu[t]) * rsqrtf(var[t] + 1e-5f) * gw[t] + bw[t];
    __syncthreads();
    float s = bo1[t];
    for (int j = 0; j < 64; j++) s += hn[j] * Wo1[j * OUTD + t];
    t1[t] = fmaxf(s, 0.f);
    __syncthreads();
    float s2 = bo2[t];
    for (int j = 0; j < 64; j++) s2 += t1[j] * Wo2[j * OUTD + t];
    out[(long)r * OUTD + t] = s2;
}

// ---------------- workspace layout (bytes, 256-aligned) ----------------
#define OFF_FV    0UL                               // VTOT*8*4   = 4 MiB
#define OFF_FE    4194304UL                         // ETOT*8*4   = 8 MiB
#define OFF_GAM   12582912UL                        // 4 MiB
#define OFF_ORDER 16777216UL                        // 256*8192*2 = 4 MiB
#define OFF_DEATH 20971520UL                        // 8192*128*2 = 2 MiB
#define OFF_CYC   23068672UL                        // 8192*256   = 2 MiB
#define OFF_H     25165824UL                        // 1024*64*4
#define OFF_MU    25427968UL
#define OFF_VAR   25428224UL

extern "C" void kernel_launch(void* const* d_in, const int* in_sizes, int n_in,
                              void* d_out, int out_size, void* d_ws, size_t ws_size,
                              hipStream_t stream) {
    (void)in_sizes; (void)n_in; (void)out_size; (void)ws_size;

    const float* x    = (const float*)d_in[0];
    const int*   eidx = (const int*)  d_in[1];
    const int*   e0   = eidx;               // edge_index[0], global node ids
    const int*   e1   = eidx + ETOT;        // edge_index[1]
    const float* Wv1 = (const float*)d_in[5],  *bv1 = (const float*)d_in[6];
    const float* Wv2 = (const float*)d_in[7],  *bv2 = (const float*)d_in[8];
    const float* We1 = (const float*)d_in[9],  *be1 = (const float*)d_in[10];
    const float* We2 = (const float*)d_in[11], *be2 = (const float*)d_in[12];
    const float* Wds0 = (const float*)d_in[13], *bds0 = (const float*)d_in[14];
    const float* Wds1 = (const float*)d_in[15], *bds1 = (const float*)d_in[16];
    const float* bn_g = (const float*)d_in[17], *bn_b = (const float*)d_in[18];
    const float* Wo1 = (const float*)d_in[19], *bo1 = (const float*)d_in[20];
    const float* Wo2 = (const float*)d_in[21], *bo2 = (const float*)d_in[22];
    float* out = (float*)d_out;

    char* ws = (char*)d_ws;
    float*          fv      = (float*)(ws + OFF_FV);
    float*          fe      = (float*)(ws + OFF_FE);
    float*          gam     = (float*)(ws + OFF_GAM);
    unsigned short* order_t = (unsigned short*)(ws + OFF_ORDER);
    short*          death   = (short*)(ws + OFF_DEATH);
    unsigned char*  cyc     = (unsigned char*)(ws + OFF_CYC);
    float*          hbuf    = (float*)(ws + OFF_H);
    float*          mu      = (float*)(ws + OFF_MU);
    float*          var     = (float*)(ws + OFF_VAR);

    // 1) filtrations (WMMA GEMMs)
    filt_kernel<false><<<VTOT / 128, 256, 0, stream>>>(x, e0, e1, Wv1, bv1, Wv2, bv2, fv);
    filt_kernel<true ><<<ETOT / 128, 256, 0, stream>>>(x, e0, e1, We1, be1, We2, be2, fe);

    // 2) gamma = min incident edge filtration
    gamma_init<<<VTOT * NFILT / 256, 256, 0, stream>>>(gam);
    gamma_scatter<<<ETOT * NFILT / 256, 256, 0, stream>>>(fe, e0, e1, (unsigned int*)gam);

    // 3) stable argsort of edge filtrations per (f, graph)
    sort_kernel<<<NPROB, 256, 0, stream>>>(fe, order_t);

    // 4) union-find persistence, one problem per lane
    uf_kernel<<<NPROB / 64, 64, 0, stream>>>(fv, e0, e1, order_t, death, cyc);

    // 5) per-graph pooled0 + pooled1 -> h
    pool_kernel<<<BG, 256, 0, stream>>>(fv, fe, gam, death, cyc,
                                        Wds0, bds0, Wds1, bds1, hbuf);

    // 6) batch stats, 7) BN + output MLP
    stats_kernel<<<1, 256, 0, stream>>>(hbuf, mu, var);
    out_kernel<<<BG, 64, 0, stream>>>(hbuf, mu, var, bn_g, bn_b, Wo1, bo1, Wo2, bo2, out);
}
